// Attention_59339268161917
// MI455X (gfx1250) — matrix operand order repaired and checked by reading, not compiled, not run on hardware
//
#include <hip/hip_runtime.h>

// Problem constants (from reference): B=2, S=2048, DIM=2048, H=16, DH=128
#define S_LEN 2048
#define DIM   2048
#define NH    16
#define DH    128

typedef __attribute__((ext_vector_type(16))) __bf16 v16bf;
typedef __attribute__((ext_vector_type(8)))  __bf16 bf16x8;
typedef __attribute__((ext_vector_type(8)))  float  v8f;

#if __has_builtin(__builtin_amdgcn_tensor_load_to_lds) && \
    __has_builtin(__builtin_amdgcn_s_wait_tensorcnt)
#define HAVE_TDM 1
#else
#define HAVE_TDM 0
#endif

__device__ __forceinline__ v8f wmma_bf16(v16bf a, v16bf b, v8f c) {
    // v_wmma_f32_16x16x32_bf16: D = A(16x32) * B(32x16) + C(16x16 f32)
    return __builtin_amdgcn_wmma_f32_16x16x32_bf16(
        false, a, false, b, (short)0, c, false, false);
}

// A-fragment (16x32 bf16, M x K): lane m = lane&15; lanes<16 hold K 0-7 & 16-23,
// lanes>=16 hold K 8-15 & 24-31 (ISA 7.12.2).
__device__ __forceinline__ v16bf ld_afrag(const __bf16* p0, int ld, int lane) {
    const __bf16* p = p0 + (lane & 15) * ld + ((lane >> 4) << 3);
    v16bf a;
#pragma unroll
    for (int i = 0; i < 8; ++i) { a[i] = p[i]; a[i + 8] = p[i + 16]; }
    return a;
}

// B-fragment (32x16 bf16, K x N) read from an [N][K]-major tile: lane n =
// lane&15; lanes<16 hold K 0-7 & 16-23, lanes>=16 hold K 8-15 & 24-31.
__device__ __forceinline__ v16bf ld_bfrag(const __bf16* p0, int ld, int lane) {
    const __bf16* p = p0 + (lane & 15) * ld + ((lane >> 4) << 3);
    v16bf b;
#pragma unroll
    for (int i = 0; i < 8; ++i) { b[i] = p[i]; b[i + 8] = p[i + 16]; }
    return b;
}

#if HAVE_TDM
typedef __attribute__((ext_vector_type(4))) unsigned int u32x4;
typedef __attribute__((ext_vector_type(8))) int          i32x8;
typedef __attribute__((ext_vector_type(4))) int          i32x4;

// Issue one TDM 2D tile load (bf16 elements): tile1 rows of tile0 contiguous
// elements, row stride stride0 (elements). D# layout per CDNA5 ISA ch.8.
// Toolchain uses the 6-arg builtin: (g0, g1, g2, g3, g_extra, cpol).
__device__ __forceinline__ void tdm_load_2d(unsigned lds_byte_addr,
                                            const __bf16* gptr,
                                            unsigned tile0, unsigned tile1,
                                            unsigned long long stride0) {
    unsigned long long ga = (unsigned long long)(uintptr_t)gptr;
    u32x4 g0;
    g0[0] = 1u;                                             // count=1 (valid), user mode
    g0[1] = lds_byte_addr;                                  // lds_addr [63:32]
    g0[2] = (unsigned)ga;                                   // global_addr [95:64]
    g0[3] = (unsigned)((ga >> 32) & 0x1FFFFFFull)           // global_addr [120:96]
            | (2u << 30);                                   // type=2 (image)
    i32x8 g1;
    g1[0] = (int)(1u << 16);                                // data_size=1 -> 2 bytes
    g1[1] = (int)((tile0 & 0xFFFFu) << 16);                 // tensor_dim0[15:0]
    g1[2] = (int)(((tile0 >> 16) & 0xFFFFu) |               // tensor_dim0[31:16]
                  ((tile1 & 0xFFFFu) << 16));               // tensor_dim1[15:0]
    g1[3] = (int)(((tile1 >> 16) & 0xFFFFu) |               // tensor_dim1[31:16]
                  ((tile0 & 0xFFFFu) << 16));               // tile_dim0
    g1[4] = (int)(tile1 & 0xFFFFu);                         // tile_dim1 (tile_dim2=0)
    g1[5] = (int)(stride0 & 0xFFFFFFFFull);                 // tensor_dim0_stride[31:0]
    g1[6] = (int)((stride0 >> 32) & 0xFFFFull);             // stride0[47:32], stride1=0
    g1[7] = 0;
    i32x4 z4 = {0, 0, 0, 0};
    i32x8 z8 = {0, 0, 0, 0, 0, 0, 0, 0};
    __builtin_amdgcn_tensor_load_to_lds(g0, g1, z4, z4, z8, 0);
}
#endif

// ---------------------------------------------------------------------------
// fp32 -> bf16 conversion (n must be a multiple of 2048)
// ---------------------------------------------------------------------------
__global__ __launch_bounds__(256) void cvt_bf16_kernel(const float* __restrict__ src,
                                                       __bf16* __restrict__ dst,
                                                       size_t n) {
    size_t i = ((size_t)blockIdx.x * blockDim.x + threadIdx.x) * 8;
    if (i + 8 > n) return;
    float4 f0 = *(const float4*)(src + i);
    float4 f1 = *(const float4*)(src + i + 4);
    bf16x8 o;
    o[0] = (__bf16)f0.x; o[1] = (__bf16)f0.y; o[2] = (__bf16)f0.z; o[3] = (__bf16)f0.w;
    o[4] = (__bf16)f1.x; o[5] = (__bf16)f1.y; o[6] = (__bf16)f1.z; o[7] = (__bf16)f1.w;
    *(bf16x8*)(dst + i) = o;
}

// ---------------------------------------------------------------------------
// GEMM: C[M=4096, N=2048] = A[4096, 2048] * W[2048, 2048]^T   (bf16 in, f32 acc)
// Block tile 128(M) x 64(N), 8 waves (4x2), wave tile 32x32, K-step 32 via LDS.
// MODE 0: store f32 row-major to outF                (final Wo projection)
// MODE 1: RoPE + store bf16 to outB as [b,h,s,d]     (Q, K)
// MODE 3: store bf16 to outB as [b,h,d,s] transposed (V; vectorized stores)
// ---------------------------------------------------------------------------
template <int MODE>
__global__ __launch_bounds__(256) void gemm_wmma_kernel(
    const __bf16* __restrict__ A, const __bf16* __restrict__ W,
    __bf16* __restrict__ outB, float* __restrict__ outF,
    const float* __restrict__ fcos, const float* __restrict__ fsin) {
    __shared__ __bf16 As[128 * 32];  // [m][k]
    __shared__ __bf16 Bs[64 * 32];   // [n][k]  (W tile, row-major along K)

    const int lane = threadIdx.x & 31;
    const int wave = threadIdx.x >> 5;
    const int bm = blockIdx.y * 128, bn = blockIdx.x * 64;
    const int wm = (wave >> 1) * 32, wn = (wave & 1) * 32;

    v8f acc[2][2] = {};

    for (int k0 = 0; k0 < DIM; k0 += 32) {
        __syncthreads();
#pragma unroll
        for (int i = 0; i < 2; ++i) {  // A tile: 128x32
            int idx = threadIdx.x * 8 + i * 2048;
            int r = idx >> 5, c = idx & 31;
            *(bf16x8*)&As[idx] = *(const bf16x8*)(A + (size_t)(bm + r) * DIM + k0 + c);
        }
        {  // W tile: 64x32
            int idx = threadIdx.x * 8;
            int r = idx >> 5, c = idx & 31;
            *(bf16x8*)&Bs[idx] = *(const bf16x8*)(W + (size_t)(bn + r) * DIM + k0 + c);
        }
        if (k0 + 32 < DIM) {  // prefetch next K-step (global_prefetch_b8)
            int idx = threadIdx.x * 8;
            __builtin_prefetch(A + (size_t)(bm + (idx >> 5)) * DIM + k0 + 32 + (idx & 31), 0, 1);
            __builtin_prefetch(A + (size_t)(bm + ((idx + 2048) >> 5)) * DIM + k0 + 32 + (idx & 31), 0, 1);
            __builtin_prefetch(W + (size_t)(bn + (idx >> 5)) * DIM + k0 + 32 + (idx & 31), 0, 1);
        }
        __syncthreads();

        v16bf a0 = ld_afrag(As + (wm + 0) * 32, 32, lane);
        v16bf a1 = ld_afrag(As + (wm + 16) * 32, 32, lane);
        v16bf b0 = ld_bfrag(Bs + (wn + 0) * 32, 32, lane);
        v16bf b1 = ld_bfrag(Bs + (wn + 16) * 32, 32, lane);
        acc[0][0] = wmma_bf16(a0, b0, acc[0][0]);
        acc[0][1] = wmma_bf16(a0, b1, acc[0][1]);
        acc[1][0] = wmma_bf16(a1, b0, acc[1][0]);
        acc[1][1] = wmma_bf16(a1, b1, acc[1][1]);
    }

    // Epilogue. C fragment: lane n = lane&15; VGPR r -> row r (lanes<16) / r+8.
    const int half8 = (lane >> 4) << 3;
    const int nn = lane & 15;
#pragma unroll
    for (int i = 0; i < 2; ++i) {
#pragma unroll
        for (int j = 0; j < 2; ++j) {
            int gmb = bm + wm + i * 16 + half8;  // first of 8 consecutive rows
            int gn = bn + wn + j * 16 + nn;
            if (MODE == 0) {
#pragma unroll
                for (int r = 0; r < 8; ++r)
                    outF[(size_t)(gmb + r) * DIM + gn] = acc[i][j][r];
            } else if (MODE == 1) {
                int h = gn >> 7, d = gn & (DH - 1);
#pragma unroll
                for (int r = 0; r < 8; ++r) {
                    int gm = gmb + r;
                    int b = gm >> 11, s = gm & (S_LEN - 1);
                    float v = acc[i][j][r];
                    // RoPE: pair (even d = real, odd d = imag) lives in lane^1
                    float partner = __shfl_xor(v, 1, 32);
                    int jj = d >> 1;
                    float c = fcos[s * (DH / 2) + jj];
                    float sn = fsin[s * (DH / 2) + jj];
                    v = ((d & 1) == 0) ? (v * c - partner * sn)
                                       : (partner * sn + v * c);
                    outB[(((size_t)b * NH + h) * S_LEN + s) * DH + d] = (__bf16)v;
                }
            } else {  // MODE 3: V transposed [b,h,d,s], 8 consecutive s per lane
                int h = gn >> 7, d = gn & (DH - 1);
                int b = gmb >> 11, s = gmb & (S_LEN - 1);
                bf16x8 o;
#pragma unroll
                for (int r = 0; r < 8; ++r) o[r] = (__bf16)acc[i][j][r];
                *(bf16x8*)(outB + (((size_t)b * NH + h) * DH + d) * S_LEN + s) = o;
            }
        }
    }
}

// ---------------------------------------------------------------------------
// Flash attention, causal. Grid: (S/128, B*H). 8 waves, 16 query rows / wave.
// Q fragments in registers; K tiles [key][d] and V tiles [d][key] double-
// buffered in LDS, filled by TDM (tensor_load_to_lds, TENSORcnt) when
// available; P transposed via per-wave LDS buffer. Output stored bf16 in the
// reference's (b,h,d,t) layout so the Wo GEMM reads it row-major.
// ---------------------------------------------------------------------------
__global__ __launch_bounds__(256) void flash_attn_kernel(
    const __bf16* __restrict__ Qb, const __bf16* __restrict__ Kb,
    const __bf16* __restrict__ Vt, __bf16* __restrict__ Ob) {
    __shared__ __bf16 Ks[2][64 * 128];   // [key][d]
    __shared__ __bf16 Vs[2][128 * 64];   // [d][key]
    __shared__ __bf16 Ps[8 * 16 * 64];   // per-wave P tile [m][key]

    const int lane = threadIdx.x & 31;
    const int wave = threadIdx.x >> 5;
    const int bh = blockIdx.y;                 // b*NH + h
    const int q0 = blockIdx.x * 128;
    const int b = bh >> 4, h = bh & 15;
    const float scale = 0.08838834764831845f;  // 1/sqrt(128)

#if HAVE_TDM
    auto tile_issue = [&](int kt, int buf) {
        tdm_load_2d((unsigned)(uintptr_t)&Ks[buf][0],
                    Kb + ((size_t)bh * S_LEN + kt * 64) * DH, 128, 64, 128);
        tdm_load_2d((unsigned)(uintptr_t)&Vs[buf][0],
                    Vt + (size_t)bh * DH * S_LEN + kt * 64, 64, 128, S_LEN);
    };
#else
    auto tile_load = [&](int kt, int buf) {
#pragma unroll
        for (int i = 0; i < 4; ++i) {
            int idx = threadIdx.x * 8 + i * 2048;
            int key = idx >> 7, d = idx & 127;
            *(bf16x8*)&Ks[buf][idx] =
                *(const bf16x8*)(Kb + ((size_t)bh * S_LEN + kt * 64 + key) * DH + d);
            int dv = idx >> 6, kv = idx & 63;
            *(bf16x8*)&Vs[buf][idx] =
                *(const bf16x8*)(Vt + ((size_t)bh * DH + dv) * S_LEN + kt * 64 + kv);
        }
    };
#endif

    // Preload Q fragments for this wave's 16 rows (4 K-steps over DH=128).
    v16bf qf[4];
    {
        int srow = q0 + wave * 16 + (lane & 15);
        const __bf16* qp = Qb + ((size_t)bh * S_LEN + srow) * DH;
#pragma unroll
        for (int ks = 0; ks < 4; ++ks) qf[ks] = ld_afrag(qp + ks * 32, 0, lane);
    }

    v8f ov[8] = {};  // O accumulators, 8 tiles over d
    float rmax[8], rsum[8];
#pragma unroll
    for (int r = 0; r < 8; ++r) { rmax[r] = -1.0e30f; rsum[r] = 0.0f; }

    const int nkt = q0 / 64 + 2;  // causal: keys up to q0+127

#if HAVE_TDM
    if (threadIdx.x < 32) tile_issue(0, 0);
#else
    tile_load(0, 0);
#endif

    for (int kt = 0; kt < nkt; ++kt) {
        const int cur = kt & 1;
        __syncthreads();  // all reads of buffer cur^1 (tile kt-1) complete
#if HAVE_TDM
        if (threadIdx.x < 32) {
            if (kt + 1 < nkt) {
                tile_issue(kt + 1, cur ^ 1);                // overlap DMA w/ compute
                __builtin_amdgcn_s_wait_tensorcnt(2);       // tile kt (2 ops) done
            } else {
                __builtin_amdgcn_s_wait_tensorcnt(0);
            }
        }
#else
        if (kt + 1 < nkt) tile_load(kt + 1, cur ^ 1);
#endif
        __syncthreads();  // tile kt visible to all waves

        // ---- scores: S = Q * K^T   (4 key-subtiles, 4 K-steps over DH)
        v8f sc[4] = {};
#pragma unroll
        for (int ks = 0; ks < 4; ++ks) {
#pragma unroll
            for (int j = 0; j < 4; ++j) {
                v16bf bk = ld_bfrag(&Ks[cur][(j * 16) * 128] + ks * 32, 128, lane);
                sc[j] = wmma_bf16(qf[ks], bk, sc[j]);
            }
        }

        // ---- scale + causal mask + online softmax
        float mnew[8];
#pragma unroll
        for (int r = 0; r < 8; ++r) {
            int qg = q0 + wave * 16 + r + ((lane >> 4) << 3);
            float mx = -3.0e38f;
#pragma unroll
            for (int j = 0; j < 4; ++j) {
                int kg = kt * 64 + j * 16 + (lane & 15);
                float v = sc[j][r] * scale + ((kg > qg) ? -1.0e9f : 0.0f);
                sc[j][r] = v;
                mx = fmaxf(mx, v);
            }
#pragma unroll
            for (int off = 8; off >= 1; off >>= 1)
                mx = fmaxf(mx, __shfl_xor(mx, off, 32));  // reduce within half-wave
            mnew[r] = fmaxf(rmax[r], mx);
        }
#pragma unroll
        for (int r = 0; r < 8; ++r) {
            float alpha = __expf(rmax[r] - mnew[r]);
            rmax[r] = mnew[r];
            float lsum = 0.0f;
#pragma unroll
            for (int j = 0; j < 4; ++j) {
                float p = __expf(sc[j][r] - mnew[r]);
                sc[j][r] = p;
                lsum += p;
            }
#pragma unroll
            for (int off = 8; off >= 1; off >>= 1) lsum += __shfl_xor(lsum, off, 32);
            rsum[r] = rsum[r] * alpha + lsum;
#pragma unroll
            for (int t = 0; t < 8; ++t) ov[t][r] *= alpha;
        }

        // ---- transpose P (C layout -> A layout) through per-wave LDS buffer
        __bf16* pw = Ps + wave * (16 * 64);
#pragma unroll
        for (int j = 0; j < 4; ++j)
#pragma unroll
            for (int r = 0; r < 8; ++r) {
                int m = r + ((lane >> 4) << 3);
                pw[m * 64 + j * 16 + (lane & 15)] = (__bf16)sc[j][r];
            }
        // same-wave DS ops are in-order; no block barrier needed

        // ---- O += P * V   (2 K-steps over 64 keys, 8 d-subtiles)
#pragma unroll
        for (int ks2 = 0; ks2 < 2; ++ks2) {
            v16bf ap = ld_afrag(pw + ks2 * 32, 64, lane);
#pragma unroll
            for (int t = 0; t < 8; ++t) {
                v16bf bv = ld_bfrag(&Vs[cur][(t * 16) * 64] + ks2 * 32, 64, lane);
                ov[t] = wmma_bf16(ap, bv, ov[t]);
            }
        }
    }

    // ---- finalize: O /= rowsum; store (b,h,d,t) with vectorized stores
    float inv[8];
#pragma unroll
    for (int r = 0; r < 8; ++r) inv[r] = 1.0f / rsum[r];
    const int tbase = q0 + wave * 16 + ((lane >> 4) << 3);
#pragma unroll
    for (int t = 0; t < 8; ++t) {
        int d = t * 16 + (lane & 15);
        bf16x8 o;
#pragma unroll
        for (int r = 0; r < 8; ++r) o[r] = (__bf16)(ov[t][r] * inv[r]);
        *(bf16x8*)(Ob + ((size_t)b * DIM + h * DH + d) * S_LEN + tbase) = o;
    }
}

// ---------------------------------------------------------------------------
extern "C" void kernel_launch(void* const* d_in, const int* in_sizes, int n_in,
                              void* d_out, int out_size, void* d_ws, size_t ws_size,
                              hipStream_t stream) {
    (void)in_sizes; (void)n_in; (void)out_size; (void)ws_size;
    const float* x  = (const float*)d_in[0];
    const float* Wq = (const float*)d_in[1];
    const float* Wk = (const float*)d_in[2];
    const float* Wv = (const float*)d_in[3];
    const float* Wo = (const float*)d_in[4];
    const float* fc = (const float*)d_in[5];
    const float* fs = (const float*)d_in[6];
    // d_in[7] = mask: unused, causal mask computed analytically
    float* out = (float*)d_out;

    const size_t nx = (size_t)2 * S_LEN * DIM;  // 8,388,608
    const size_t nw = (size_t)DIM * DIM;        // 4,194,304

    // Workspace layout (bf16), ~117 MB total
    __bf16* xb = (__bf16*)d_ws;
    __bf16* wq = xb + nx;
    __bf16* wk = wq + nw;
    __bf16* wv = wk + nw;
    __bf16* wo = wv + nw;
    __bf16* qb = wo + nw;  // [b,h,s,d]
    __bf16* kb = qb + nx;  // [b,h,s,d]
    __bf16* vt = kb + nx;  // [b,h,d,s]  (transposed V)
    __bf16* ob = vt + nx;  // [b,h,d,t] == reference's reshaped (b,s,dim) matrix

    cvt_bf16_kernel<<<nx / 2048, 256, 0, stream>>>(x, xb, nx);
    cvt_bf16_kernel<<<nw / 2048, 256, 0, stream>>>(Wq, wq, nw);
    cvt_bf16_kernel<<<nw / 2048, 256, 0, stream>>>(Wk, wk, nw);
    cvt_bf16_kernel<<<nw / 2048, 256, 0, stream>>>(Wv, wv, nw);
    cvt_bf16_kernel<<<nw / 2048, 256, 0, stream>>>(Wo, wo, nw);

    dim3 gg(DIM / 64, (2 * S_LEN) / 128);  // (32, 32)
    gemm_wmma_kernel<1><<<gg, 256, 0, stream>>>(xb, wq, qb, nullptr, fc, fs);
    gemm_wmma_kernel<1><<<gg, 256, 0, stream>>>(xb, wk, kb, nullptr, fc, fs);
    gemm_wmma_kernel<3><<<gg, 256, 0, stream>>>(xb, wv, vt, nullptr, nullptr, nullptr);

    flash_attn_kernel<<<dim3(S_LEN / 128, 2 * NH), 256, 0, stream>>>(qb, kb, vt, ob);

    gemm_wmma_kernel<0><<<gg, 256, 0, stream>>>(ob, wo, nullptr, out, nullptr, nullptr);
}
